// Attention_87771951661841
// MI455X (gfx1250) — matrix operand order repaired and checked by reading, not compiled, not run on hardware
//
#include <hip/hip_runtime.h>
#include <hip/hip_bf16.h>
#include <math.h>

typedef unsigned short u16;
typedef __attribute__((ext_vector_type(16))) __bf16 v16bf;
typedef __attribute__((ext_vector_type(8)))  float  v8f;
typedef __attribute__((ext_vector_type(4)))  unsigned u32x4;
typedef __attribute__((ext_vector_type(2)))  unsigned u32x2;
typedef __attribute__((ext_vector_type(4)))  float    f32x4;

#define B_SZ   4
#define S_LEN  2048
#define DMODEL 1024
#define NHEAD  16
#define HD     64
#define MTOT   (B_SZ * S_LEN)   // 8192

__device__ __forceinline__ u16 f2bf(float f) {
  unsigned u = __float_as_uint(f);
  return (u16)((u + 0x7FFFu + ((u >> 16) & 1u)) >> 16);
}

union ABFrag { u32x4 q[2]; v16bf v; };

// A fragment (16x32 bf16): lane r=l&15 holds row r; chunks at K = hi*8 and 16+hi*8.
__device__ __forceinline__ v16bf lds_a_frag(const u16* tile, int lane, int strideHalves) {
  const int r = lane & 15, hi = (lane >> 4) & 1;
  const u16* row = tile + r * strideHalves;
  ABFrag f;
  f.q[0] = *(const u32x4*)(row + hi * 8);
  f.q[1] = *(const u32x4*)(row + 16 + hi * 8);
  return f.v;
}

// B fragment (32x16 bf16) from a [N][K]-transposed tile: lane holds col c=col0+(l&15),
// chunks at K = hi*8 and 16+hi*8.
__device__ __forceinline__ v16bf lds_b_frag(const u16* tile, int col0, int lane, int strideHalves) {
  const int c = col0 + (lane & 15), hi = (lane >> 4) & 1;
  const u16* p = tile + c * strideHalves + hi * 8;
  ABFrag f;
  f.q[0] = *(const u32x4*)(p);
  f.q[1] = *(const u32x4*)(p + 16);
  return f.v;
}

__device__ __forceinline__ v8f wmma_bf16(v16bf a, v16bf b, v8f c) {
  return __builtin_amdgcn_wmma_f32_16x16x32_bf16(false, a, false, b, (short)0, c, false, false);
}

// Async global->LDS copies (ASYNCcnt-tracked, CDNA5). lds = 32-bit LDS byte address,
// g = 64-bit global address; the immediate offset is added to BOTH addresses.
__device__ __forceinline__ void async_cp32(unsigned lds, const void* g) {
  asm volatile("global_load_async_to_lds_b128 %0, %1, off\n\t"
               "global_load_async_to_lds_b128 %0, %1, off offset:16"
               :: "v"(lds), "v"(g) : "memory");
}
__device__ __forceinline__ void async_cp64(unsigned lds, const void* g) {
  asm volatile("global_load_async_to_lds_b128 %0, %1, off\n\t"
               "global_load_async_to_lds_b128 %0, %1, off offset:16\n\t"
               "global_load_async_to_lds_b128 %0, %1, off offset:32\n\t"
               "global_load_async_to_lds_b128 %0, %1, off offset:48"
               :: "v"(lds), "v"(g) : "memory");
}
__device__ __forceinline__ void async_wait0() {
  asm volatile("s_wait_asynccnt 0x0" ::: "memory");
}
__device__ __forceinline__ unsigned lds_addr(const void* p) {
  return (unsigned)(size_t)p;  // low 32 bits of a flat shared address = LDS byte offset
}

// ---------------- fp32 -> bf16 conversion ----------------
__global__ void cvt_f32_bf16(const float* __restrict__ in, u16* __restrict__ out, int n) {
  int i = (blockIdx.x * blockDim.x + threadIdx.x) * 4;
  if (i >= n) return;
  f32x4 f = *(const f32x4*)(in + i);
  u32x2 d;
  d[0] = (unsigned)f2bf(f[0]) | ((unsigned)f2bf(f[1]) << 16);
  d[1] = (unsigned)f2bf(f[2]) | ((unsigned)f2bf(f[3]) << 16);
  *(u32x2*)(out + i) = d;
}

// ---------------- bf16 GEMM: C = A[M,K] @ B[K,N] + bias ----------------
// Block 256 threads = 8 waves as 4(M) x 2(N). Tile: 128(M) x 128(N), K-step 32.
// Each wave: 32 rows x 64 cols -> 2 A-frags, 4 B-frags, 8 WMMAs per K-step.
template <bool OUT_F32>
__launch_bounds__(256)
__global__ void gemm_bf16_kernel(const u16* __restrict__ A, const u16* __restrict__ Bm,
                                 const float* __restrict__ bias, void* __restrict__ Cout,
                                 int M, int N, int K) {
  __shared__ __attribute__((aligned(16))) u16 sA[128 * 32]; // [M][K]
  __shared__ __attribute__((aligned(16))) u16 sB[128 * 32]; // [N][K] (transposed)
  const int tid  = threadIdx.x;
  const int wave = tid >> 5, lane = tid & 31;
  const int waveM = wave >> 1, waveN = wave & 1;
  const int r = lane & 15, hi = (lane >> 4) & 1;
  const int m0 = blockIdx.y * 128, n0 = blockIdx.x * 128;

  v8f acc[2][4];
  #pragma unroll
  for (int s = 0; s < 2; ++s)
    #pragma unroll
    for (int n = 0; n < 4; ++n)
      #pragma unroll
      for (int g = 0; g < 8; ++g) acc[s][n][g] = 0.f;

  const int rrA = tid >> 1, hhA = tid & 1;          // A tile: 32B per thread
  const unsigned ldsA = lds_addr(&sA[rrA * 32 + hhA * 16]);
  const int kkB = tid >> 3, ncB = tid & 7;          // B tile: 16 cols per thread

  for (int k0 = 0; k0 < K; k0 += 32) {
    // A tile: async copy, layout-preserving
    async_cp32(ldsA, A + (size_t)(m0 + rrA) * K + k0 + hhA * 16);
    // B tile: load + transpose into [N][K]
    {
      const u16* g = Bm + (size_t)(k0 + kkB) * N + n0 + ncB * 16;
      u32x4 d0 = *(const u32x4*)(g);
      u32x4 d1 = *(const u32x4*)(g + 8);
      const u16* e0 = (const u16*)&d0;
      const u16* e1 = (const u16*)&d1;
      #pragma unroll
      for (int i = 0; i < 8; ++i) {
        sB[(ncB * 16 + i) * 32 + kkB]     = e0[i];
        sB[(ncB * 16 + 8 + i) * 32 + kkB] = e1[i];
      }
    }
    async_wait0();
    __syncthreads();

    v16bf af0 = lds_a_frag(&sA[(waveM * 32) * 32],      lane, 32);
    v16bf af1 = lds_a_frag(&sA[(waveM * 32 + 16) * 32], lane, 32);
    const u16* sBw = sB + (waveN * 64) * 32;
    #pragma unroll
    for (int n = 0; n < 4; ++n) {
      v16bf bf = lds_b_frag(sBw, n * 16, lane, 32);
      acc[0][n] = wmma_bf16(af0, bf, acc[0][n]);
      acc[1][n] = wmma_bf16(af1, bf, acc[1][n]);
    }
    __syncthreads();
  }

  #pragma unroll
  for (int s = 0; s < 2; ++s)
    #pragma unroll
    for (int n = 0; n < 4; ++n) {
      int c = n0 + waveN * 64 + n * 16 + r;
      float bv = bias[c];
      #pragma unroll
      for (int g = 0; g < 8; ++g) {
        size_t row = (size_t)(m0 + waveM * 32 + s * 16 + hi * 8 + g);
        float vo = acc[s][n][g] + bv;
        if constexpr (OUT_F32) ((float*)Cout)[row * (size_t)N + c] = vo;
        else                   ((u16*)Cout)[row * (size_t)N + c]  = f2bf(vo);
      }
    }
}

// ---------------- flash attention ----------------
// grid: (S/64, B*H); block 128 (4 waves, 16 query rows each).
__launch_bounds__(128)
__global__ void flash_attn_kernel(const u16* __restrict__ qkv, u16* __restrict__ attnout) {
  __shared__ __attribute__((aligned(16))) u16 kbuf[64 * HD];     // [key][hd]
  __shared__ __attribute__((aligned(16))) u16 vbuf[HD * 64];     // [hd][key] (transposed)
  __shared__ __attribute__((aligned(16))) u16 pscr[4][16 * 64];  // per-wave P scratch

  const int tid  = threadIdx.x;
  const int wave = tid >> 5, lane = tid & 31;
  const int r = lane & 15, hi = (lane >> 4) & 1;
  const int bh = blockIdx.y, b = bh >> 4, h = bh & 15;
  const int q0 = blockIdx.x * 64;
  const size_t rs = 3 * DMODEL; // qkv row stride in halves
  const u16* base = qkv + (size_t)b * S_LEN * rs;

  // Persistent Q fragments (A-layout, K = hd = 64 -> two frags)
  ABFrag qf0, qf1;
  {
    const u16* qp = base + (size_t)(q0 + wave * 16 + r) * rs + h * HD;
    qf0.q[0] = *(const u32x4*)(qp + hi * 8);
    qf0.q[1] = *(const u32x4*)(qp + 16 + hi * 8);
    qf1.q[0] = *(const u32x4*)(qp + 32 + hi * 8);
    qf1.q[1] = *(const u32x4*)(qp + 48 + hi * 8);
  }

  v8f o[4];
  float mrow[8], lsum[8];
  #pragma unroll
  for (int n = 0; n < 4; ++n)
    #pragma unroll
    for (int g = 0; g < 8; ++g) o[n][g] = 0.f;
  #pragma unroll
  for (int g = 0; g < 8; ++g) { mrow[g] = -__builtin_inff(); lsum[g] = 0.f; }

  const int kr = tid >> 1, hh = tid & 1;            // tile loaders: 64B per thread
  const unsigned ldsK = lds_addr(&kbuf[kr * HD + hh * 32]);

  for (int kt = 0; kt < S_LEN; kt += 64) {
    // K tile -> kbuf [key][hd]: async copy, layout-preserving
    async_cp64(ldsK, base + (size_t)(kt + kr) * rs + DMODEL + h * HD + hh * 32);
    // V tile -> vbuf [hd][key]: load + transpose on store
    {
      const u16* g = base + (size_t)(kt + kr) * rs + 2 * DMODEL + h * HD + hh * 32;
      u32x4 d[4];
      d[0] = *(const u32x4*)(g);
      d[1] = *(const u32x4*)(g + 8);
      d[2] = *(const u32x4*)(g + 16);
      d[3] = *(const u32x4*)(g + 24);
      const u16* e = (const u16*)d;
      #pragma unroll
      for (int i = 0; i < 32; ++i) vbuf[(hh * 32 + i) * 64 + kr] = e[i];
    }
    async_wait0();
    __syncthreads();

    // scores S = scale * Q @ K^T  (four 16x16 C tiles, K-dim = hd = 64)
    v8f st[4];
    #pragma unroll
    for (int n = 0; n < 4; ++n) {
      v8f s;
      #pragma unroll
      for (int g = 0; g < 8; ++g) s[g] = 0.f;
      v16bf b0 = lds_b_frag(kbuf,      n * 16, lane, HD); // hd 0..31
      v16bf b1 = lds_b_frag(kbuf + 32, n * 16, lane, HD); // hd 32..63
      s = wmma_bf16(qf0.v, b0, s);
      s = wmma_bf16(qf1.v, b1, s);
      #pragma unroll
      for (int g = 0; g < 8; ++g) s[g] *= 0.125f; // 1/sqrt(64)
      st[n] = s;
    }

    // online softmax: per C-row (row = hi*8+g) reduce across the 16-lane half
    #pragma unroll
    for (int g = 0; g < 8; ++g) {
      float x = st[0][g];
      x = fmaxf(x, st[1][g]); x = fmaxf(x, st[2][g]); x = fmaxf(x, st[3][g]);
      x = fmaxf(x, __shfl_xor(x, 1, 32));
      x = fmaxf(x, __shfl_xor(x, 2, 32));
      x = fmaxf(x, __shfl_xor(x, 4, 32));
      x = fmaxf(x, __shfl_xor(x, 8, 32));
      float mnew  = fmaxf(mrow[g], x);
      float alpha = __expf(mrow[g] - mnew);
      mrow[g] = mnew;
      float rsum = 0.f;
      #pragma unroll
      for (int n = 0; n < 4; ++n) {
        float p = __expf(st[n][g] - mnew);
        st[n][g] = p;
        rsum += p;
      }
      rsum += __shfl_xor(rsum, 1, 32);
      rsum += __shfl_xor(rsum, 2, 32);
      rsum += __shfl_xor(rsum, 4, 32);
      rsum += __shfl_xor(rsum, 8, 32);
      lsum[g] = lsum[g] * alpha + rsum;
      #pragma unroll
      for (int n = 0; n < 4; ++n) o[n][g] *= alpha;
    }

    // Re-layout P (C layout) -> A layout via per-wave LDS scratch
    u16* ps = pscr[wave];
    #pragma unroll
    for (int n = 0; n < 4; ++n)
      #pragma unroll
      for (int g = 0; g < 8; ++g)
        ps[(hi * 8 + g) * 64 + n * 16 + r] = f2bf(st[n][g]);
    v16bf pf0 = lds_a_frag(ps,      lane, 64); // keys 0..31
    v16bf pf1 = lds_a_frag(ps + 32, lane, 64); // keys 32..63

    // O += P @ V
    #pragma unroll
    for (int n = 0; n < 4; ++n) {
      v16bf v0 = lds_b_frag(vbuf,      n * 16, lane, 64); // keys 0..31
      v16bf v1 = lds_b_frag(vbuf + 32, n * 16, lane, 64); // keys 32..63
      o[n] = wmma_bf16(pf0, v0, o[n]);
      o[n] = wmma_bf16(pf1, v1, o[n]);
    }
    __syncthreads();
  }

  // epilogue: normalize and store bf16 to [B*S][DMODEL]
  #pragma unroll
  for (int n = 0; n < 4; ++n) {
    int c = h * HD + n * 16 + r;
    #pragma unroll
    for (int g = 0; g < 8; ++g) {
      size_t row = (size_t)b * S_LEN + q0 + wave * 16 + hi * 8 + g;
      attnout[row * DMODEL + c] = f2bf(o[n][g] / lsum[g]);
    }
  }
}

// ---------------- launcher ----------------
extern "C" void kernel_launch(void* const* d_in, const int* in_sizes, int n_in,
                              void* d_out, int out_size, void* d_ws, size_t ws_size,
                              hipStream_t stream) {
  const float* x     = (const float*)d_in[0];
  const float* w_qkv = (const float*)d_in[1];
  const float* b_qkv = (const float*)d_in[2];
  const float* w_fc  = (const float*)d_in[3];
  const float* b_fc  = (const float*)d_in[4];

  u16* xb    = (u16*)d_ws;                               // 8192x1024 bf16
  u16* wqkvb = xb    + (size_t)MTOT * DMODEL;            // 1024x3072 bf16
  u16* wfcb  = wqkvb + (size_t)DMODEL * 3 * DMODEL;      // 1024x1024 bf16
  u16* qkvb  = wfcb  + (size_t)DMODEL * DMODEL;          // 8192x3072 bf16
  u16* attnb = qkvb  + (size_t)MTOT * 3 * DMODEL;        // 8192x1024 bf16

  const int nx  = MTOT * DMODEL;
  const int nwq = DMODEL * 3 * DMODEL;
  const int nwf = DMODEL * DMODEL;
  cvt_f32_bf16<<<nx  / 1024, 256, 0, stream>>>(x,     xb,    nx);
  cvt_f32_bf16<<<nwq / 1024, 256, 0, stream>>>(w_qkv, wqkvb, nwq);
  cvt_f32_bf16<<<nwf / 1024, 256, 0, stream>>>(w_fc,  wfcb,  nwf);

  dim3 g1(3 * DMODEL / 128, MTOT / 128);
  gemm_bf16_kernel<false><<<g1, 256, 0, stream>>>(xb, wqkvb, b_qkv, qkvb,
                                                  MTOT, 3 * DMODEL, DMODEL);

  dim3 g2(S_LEN / 64, B_SZ * NHEAD);
  flash_attn_kernel<<<g2, 128, 0, stream>>>(qkvb, attnb);

  dim3 g3(DMODEL / 128, MTOT / 128);
  gemm_bf16_kernel<true><<<g3, 256, 0, stream>>>(attnb, wfcb, b_fc, d_out,
                                                 MTOT, DMODEL, DMODEL);
}
